// KNNConLoss_78331613544657
// MI455X (gfx1250) — matrix operand order repaired and checked by reading, not compiled, not run on hardware
//
#include <hip/hip_runtime.h>
#include <stdint.h>

// ---------------------------------------------------------------------------
// KNN contrastive loss for MI455X (gfx1250, wave32).
//  * Gram matrix via split-precision bf16 WMMA (hi*hi + hi*lo + lo*hi)
//  * B tiles staged in LDS by the Tensor Data Mover (double buffered,
//    TENSORcnt-synchronized, LDS padding to avoid bank conflicts)
//  * fused exp / mask / row-sum / register-resident top-10
//  * tiny finalize kernel -> 3 scalars (loss, pos_mean, neg_mean)
// ---------------------------------------------------------------------------

typedef __attribute__((ext_vector_type(16))) __bf16 v16bf;
typedef __attribute__((ext_vector_type(8)))  float  v8f;
typedef __attribute__((ext_vector_type(4)))  unsigned int v4u;
typedef __attribute__((ext_vector_type(8)))  int    v8i;
typedef __attribute__((ext_vector_type(4)))  int    v4i;

#define B_ROWS 2048
#define NROWS  4096
#define DIM    256
#define TOPKN  10
#define INV_T  20.0f

// LDS stage: 16 columns x 256 K of bf16 = 512B/col, padded +16B per 512B by
// the TDM -> column stride 528B (264 uint16 elements, 132 DWORDs -> lanes hit
// distinct banks on ds_load_b128 fragment reads).
#define COL_STRIDE_E 264              // uint16 elements per staged column
#define STAGE_E (16 * COL_STRIDE_E)   // 4224 elements = 8448 bytes per array

__device__ __forceinline__ uint16_t f32_to_bf16_rne(float x) {
  uint32_t u = __float_as_uint(x);
  uint32_t r = u + 0x7FFFu + ((u >> 16) & 1u);
  return (uint16_t)(r >> 16);
}
__device__ __forceinline__ float bf16_bits_to_f32(uint16_t h) {
  return __uint_as_float(((uint32_t)h) << 16);
}

// --------------------------- kernel 1: split convert -----------------------
__global__ __launch_bounds__(256)
void convert_split_bf16(const float* __restrict__ f1, const float* __restrict__ f2,
                        uint16_t* __restrict__ fh, uint16_t* __restrict__ fl) {
  int idx = blockIdx.x * blockDim.x + threadIdx.x;
  const int half = B_ROWS * DIM;
  if (idx >= 2 * half) return;
  float x = (idx < half) ? f1[idx] : f2[idx - half];
  uint16_t h = f32_to_bf16_rne(x);
  float hf = bf16_bits_to_f32(h);
  fh[idx] = h;
  fl[idx] = f32_to_bf16_rne(x - hf);
}

// --------------------------- kernel 2: exact positive dots ------------------
__global__ __launch_bounds__(256)
void pos_kernel(const float* __restrict__ f1, const float* __restrict__ f2,
                float* __restrict__ pos) {
  const int wave = threadIdx.x >> 5;
  const int lane = threadIdx.x & 31;
  const int row  = blockIdx.x * 8 + wave;
  float s = 0.0f;
  for (int k = lane; k < DIM; k += 32)
    s += f1[(size_t)row * DIM + k] * f2[(size_t)row * DIM + k];
  s += __shfl_xor(s, 1, 32);
  s += __shfl_xor(s, 2, 32);
  s += __shfl_xor(s, 4, 32);
  s += __shfl_xor(s, 8, 32);
  s += __shfl_xor(s, 16, 32);
  if (lane == 0) pos[row] = __expf(s * INV_T);
}

// --------------------------- TDM 8KB row-block load -------------------------
// Contiguous 16 feature rows (16*512B) -> LDS with 16B padding every 512B.
// Descriptor per CDNA5 ISA; 6-arg builtin on this
// toolchain (trailing descriptor groups unused for a 1-D tile -> zero-filled).
__device__ __forceinline__ void tdm_load_tile(const uint16_t* gp, uint32_t lds_off) {
  uint64_t ga = (uint64_t)(uintptr_t)gp;
  v4u g0;
  g0.x = 1u;                                   // count=1, user descriptor
  g0.y = lds_off;                              // lds_addr (bytes)
  g0.z = (uint32_t)ga;                         // global_addr[31:0]
  g0.w = ((uint32_t)(ga >> 32) & 0x01FFFFFFu)  // global_addr[56:32]
         | (2u << 30);                         // type=2 ("image")
  // data_size=2B (code 1), pad_enable, pad_interval code 6 (=128 DWORDs=512B),
  // pad_amount code 3 (=4 DWORDs=16B); 1-D tile of 4096 elements.
  v8i g1;
  g1.s0 = (int)((1u << 16) | (1u << 20) | (6u << 22) | (3u << 25));
  g1.s1 = (int)(4096u << 16);   // tensor_dim0[15:0] = 4096
  g1.s2 = (int)(1u << 16);      // tensor_dim1[15:0] = 1
  g1.s3 = (int)(4096u << 16);   // tile_dim0 = 4096
  g1.s4 = 1;                    // tile_dim1 = 1
  g1.s5 = 4096;                 // tensor_dim0_stride = 4096
  g1.s6 = 0;
  g1.s7 = 0;
  v4i z4 = {0, 0, 0, 0};
  v8i z8 = {0, 0, 0, 0, 0, 0, 0, 0};
  __builtin_amdgcn_tensor_load_to_lds(g0, g1, z4, z4, z8, 0);
}

// --------------------------- kernel 3: gram + exp + mask + topk -------------
union FragA { v16bf v; uint32_t u[8]; };
union FragL { v16bf v; uint4 q[2]; };
union AccU  { v8f v; float f[8]; };

__global__ __launch_bounds__(128)
void gram_topk_kernel(const uint16_t* __restrict__ fh, const uint16_t* __restrict__ fl,
                      float* __restrict__ rowSum, float* __restrict__ top10) {
  // double-buffered B stage: [buf][hi/lo][16 cols * 264 elem]  (~33 KB)
  __shared__ __align__(16) uint16_t ldsB[2][2][STAGE_E];

  const int tid   = threadIdx.x;
  const int wave  = tid >> 5;
  const int lane  = tid & 31;
  const int halfw = lane >> 4;      // which 16-lane half of the wave
  const int n     = lane & 15;
  const int R0    = (blockIdx.x * 4 + wave) * 16;   // this wave's 16-row block

  // Preload A fragments (16-bit A 16x32 layout, ISA 7.12.2):
  // lanes 0-15: M=lane, K pairs {0,2,4,6,16,18,20,22}; lanes 16-31: same M, K+8.
  FragA aHi[8], aLo[8];
  {
    const uint16_t* ph = fh + (size_t)(R0 + n) * DIM;
    const uint16_t* pl = fl + (size_t)(R0 + n) * DIM;
#pragma unroll
    for (int kt = 0; kt < 8; ++kt) {
#pragma unroll
      for (int i = 0; i < 8; ++i) {
        int kb = kt * 32 + ((i < 4) ? 0 : 16) + halfw * 8 + (i & 3) * 2;
        aHi[kt].u[i] = *(const uint32_t*)(ph + kb);
        aLo[kt].u[i] = *(const uint32_t*)(pl + kb);
      }
    }
  }

  float rsum[8];
  float topl[8][10];                 // register-resident ascending top-10 lists
#pragma unroll
  for (int r = 0; r < 8; ++r) {
    rsum[r] = 0.0f;
#pragma unroll
    for (int q = 0; q < TOPKN; ++q) topl[r][q] = 0.0f;
  }

  // Prologue: wave 0 kicks off the TDM load of column tile 0.
  if (wave == 0) {
    tdm_load_tile(fh, (uint32_t)(uintptr_t)&ldsB[0][0][0]);
    tdm_load_tile(fl, (uint32_t)(uintptr_t)&ldsB[0][1][0]);
  }

  for (int jt = 0; jt < NROWS / 16; ++jt) {
    const int cur = jt & 1;
    if (wave == 0) __builtin_amdgcn_s_wait_tensorcnt(0);
    __syncthreads();                 // buf[cur] ready; prev compute done
    if (wave == 0 && jt + 1 < NROWS / 16) {
      const uint16_t* gh = fh + (size_t)(jt + 1) * 16 * DIM;
      const uint16_t* gl = fl + (size_t)(jt + 1) * 16 * DIM;
      tdm_load_tile(gh, (uint32_t)(uintptr_t)&ldsB[cur ^ 1][0][0]);
      tdm_load_tile(gl, (uint32_t)(uintptr_t)&ldsB[cur ^ 1][1][0]);
    }

    const int C0 = jt * 16;
    const uint16_t* hiB = &ldsB[cur][0][0];
    const uint16_t* loB = &ldsB[cur][1][0];

    AccU acc = {};
#pragma unroll
    for (int kt = 0; kt < 8; ++kt) {
      // B fragments (16-bit B 32x16): lane holds column n; K 0-7 and 16-23
      // (+8 for upper half-wave), matching A -> 2x ds_load_b128.
      FragL bHi, bLo;
      const uint4* ph4 = (const uint4*)(hiB + n * COL_STRIDE_E + kt * 32 + halfw * 8);
      const uint4* pl4 = (const uint4*)(loB + n * COL_STRIDE_E + kt * 32 + halfw * 8);
      bHi.q[0] = ph4[0]; bHi.q[1] = ph4[2];
      bLo.q[0] = pl4[0]; bLo.q[1] = pl4[2];
      acc.v = __builtin_amdgcn_wmma_f32_16x16x32_bf16(false, aHi[kt].v, false, bHi.v,
                                                      (short)0, acc.v, false, false);
      acc.v = __builtin_amdgcn_wmma_f32_16x16x32_bf16(false, aHi[kt].v, false, bLo.v,
                                                      (short)0, acc.v, false, false);
      acc.v = __builtin_amdgcn_wmma_f32_16x16x32_bf16(false, aLo[kt].v, false, bHi.v,
                                                      (short)0, acc.v, false, false);
    }

    // C/D layout: VGPR r -> row (r + 8*halfw), lane n -> column n.
#pragma unroll
    for (int r = 0; r < 8; ++r) {
      const int row = R0 + r + 8 * halfw;
      const int col = C0 + n;
      float v = __expf(acc.f[r] * INV_T);
      if (((row ^ col) & (B_ROWS - 1)) == 0) v = 0.0f;   // masked pair columns
      rsum[r] += v;
      if (v > topl[r][0]) {          // insert into ascending top-10 (registers)
        topl[r][0] = v;
#pragma unroll
        for (int q = 0; q < TOPKN - 1; ++q)
          if (topl[r][q] > topl[r][q + 1]) {
            float t = topl[r][q]; topl[r][q] = topl[r][q + 1]; topl[r][q + 1] = t;
          }
      }
    }
  }

  // Row sums: reduce across the 16 lanes of each half-wave.
#pragma unroll
  for (int r = 0; r < 8; ++r) {
    float s = rsum[r];
    s += __shfl_xor(s, 1, 16);
    s += __shfl_xor(s, 2, 16);
    s += __shfl_xor(s, 4, 16);
    s += __shfl_xor(s, 8, 16);
    if (n == 0) rowSum[R0 + r + 8 * halfw] = s;
  }

  // Merge: 16 lane-lists of 10 per row -> global top-10 via repeated wave-max.
#pragma unroll
  for (int r = 0; r < 8; ++r) {
    const int row = R0 + r + 8 * halfw;
    for (int it = 0; it < TOPKN; ++it) {
      float lmax = topl[r][0];
#pragma unroll
      for (int q = 1; q < TOPKN; ++q) lmax = fmaxf(lmax, topl[r][q]);
      float gmax = lmax;
      gmax = fmaxf(gmax, __shfl_xor(gmax, 1, 16));
      gmax = fmaxf(gmax, __shfl_xor(gmax, 2, 16));
      gmax = fmaxf(gmax, __shfl_xor(gmax, 4, 16));
      gmax = fmaxf(gmax, __shfl_xor(gmax, 8, 16));
      unsigned long long bal = __ballot(lmax == gmax);
      unsigned long long grp = 0xFFFFull << (halfw * 16);
      int first = __ffsll(bal & grp) - 1;
      if (lane == first) {           // unique lane removes one instance
        bool done = false;
#pragma unroll
        for (int q = 0; q < TOPKN; ++q)
          if (!done && topl[r][q] == gmax) { topl[r][q] = -1.0f; done = true; }
      }
      if (n == 0) top10[(size_t)row * TOPKN + it] = gmax;
    }
  }
}

// --------------------------- kernel 4: finalize -----------------------------
__global__ __launch_bounds__(256)
void finalize_kernel(const float* __restrict__ rowSum, const float* __restrict__ top10,
                     const float* __restrict__ pos, float* __restrict__ out) {
  __shared__ float redL[256], redP[256], redN[256];
  float L = 0.0f, P = 0.0f, NG = 0.0f;
  for (int i = threadIdx.x; i < NROWS; i += 256) {
    float t[TOPKN]; float ts = 0.0f;
#pragma unroll
    for (int q = 0; q < TOPKN; ++q) { t[q] = top10[(size_t)i * TOPKN + q]; ts += t[q]; }
    float Ng = rowSum[i] - ts;               // sum of negatives
    float p0 = pos[i & (B_ROWS - 1)];
    P += p0 + ts;
    NG += Ng;
    L += -__logf(p0 / (Ng + p0));
#pragma unroll
    for (int q = 0; q < TOPKN; ++q) L += -__logf(t[q] / (Ng + t[q]));
  }
  redL[threadIdx.x] = L; redP[threadIdx.x] = P; redN[threadIdx.x] = NG;
  __syncthreads();
  for (int s = 128; s > 0; s >>= 1) {
    if (threadIdx.x < s) {
      redL[threadIdx.x] += redL[threadIdx.x + s];
      redP[threadIdx.x] += redP[threadIdx.x + s];
      redN[threadIdx.x] += redN[threadIdx.x + s];
    }
    __syncthreads();
  }
  if (threadIdx.x == 0) {
    const float denomPos = (float)NROWS * (float)(TOPKN + 1);
    out[0] = redL[0] / denomPos;                                    // loss
    out[1] = redP[0] / denomPos;                                    // pos_mean
    out[2] = redN[0] / ((float)NROWS * (float)(NROWS - 2 - TOPKN)); // neg_mean
  }
}

// --------------------------- launcher ---------------------------------------
extern "C" void kernel_launch(void* const* d_in, const int* in_sizes, int n_in,
                              void* d_out, int out_size, void* d_ws, size_t ws_size,
                              hipStream_t stream) {
  const float* f1 = (const float*)d_in[0];
  const float* f2 = (const float*)d_in[1];
  float* out = (float*)d_out;

  char* ws = (char*)d_ws;
  uint16_t* fh   = (uint16_t*)(ws);                                  // 2 MB
  uint16_t* fl   = (uint16_t*)(ws + (size_t)NROWS * DIM * 2);        // 2 MB
  float* pos     = (float*)(ws + (size_t)NROWS * DIM * 4);           // 8 KB
  float* rowSum  = pos + B_ROWS;                                     // 16 KB
  float* top10   = rowSum + NROWS;                                   // 160 KB

  convert_split_bf16<<<(NROWS * DIM + 255) / 256, 256, 0, stream>>>(f1, f2, fh, fl);
  pos_kernel<<<B_ROWS / 8, 256, 0, stream>>>(f1, f2, pos);
  gram_topk_kernel<<<NROWS / 64, 128, 0, stream>>>(fh, fl, rowSum, top10);
  finalize_kernel<<<1, 256, 0, stream>>>(rowSum, top10, pos, out);
}
